// Multi_Head_59880434041483
// MI455X (gfx1250) — matrix operand order repaired and checked by reading, not compiled, not run on hardware
//
#include <hip/hip_runtime.h>
#include <hip/hip_bf16.h>
#include <math.h>

#define BB 2
#define SS 2048
#define DD 1024
#define HH 16
#define DVV 64
#define LNEPS 1e-6f

typedef __attribute__((ext_vector_type(16))) __bf16 v16bf;
typedef __attribute__((ext_vector_type(8)))  float  v8f;

union BF16x16 { v16bf v; uint4 q[2]; };

__device__ __forceinline__ v8f wmma_bf16(v16bf a, v16bf b, v8f c) {
    return __builtin_amdgcn_wmma_f32_16x16x32_bf16(false, a, false, b, (short)0, c, false, false);
}

// ---------------------------------------------------------------- convert f32 -> bf16 (8 elems/thread)
__global__ __launch_bounds__(256)
void cvt_bf16_v(const float4* __restrict__ src, uint4* __restrict__ dst, int n8) {
    int i = blockIdx.x * 256 + threadIdx.x;
    if (i < n8) {
        float4 x = src[2 * i], y = src[2 * i + 1];
        union { __bf16 h[8]; uint4 u; } o;
        o.h[0] = (__bf16)x.x; o.h[1] = (__bf16)x.y; o.h[2] = (__bf16)x.z; o.h[3] = (__bf16)x.w;
        o.h[4] = (__bf16)y.x; o.h[5] = (__bf16)y.y; o.h[6] = (__bf16)y.z; o.h[7] = (__bf16)y.w;
        dst[i] = o.u;
    }
}

// ---------------------------------------------------------------- projection GEMM
// One wave computes a 32x64 tile of C = X(M x 1024 bf16) * W^T (+bias), W row-major (N,1024).
// 8 WMMAs per k-step from 12 b128 loads (1.5 loads/WMMA).
// mode 0: Q  -> bf16 head-major [bh][s][dv], value = (acc+bias)*scale
// mode 1: K  -> bf16 head-major [bh][s][dv]
// mode 2: V  -> bf16 transposed head-major [bh][dv][s]
// mode 3: O  -> f32 row-major, += residual
__global__ __launch_bounds__(32)
void proj_gemm(const __bf16* __restrict__ X, const __bf16* __restrict__ W,
               const float* __restrict__ bias, float scale, int mode,
               __bf16* __restrict__ obf, float* __restrict__ of,
               const float* __restrict__ resid)
{
    const int lane  = threadIdx.x;
    const int nbase = blockIdx.x * 64;
    const int mbase = blockIdx.y * 32;
    const int sel8  = (lane < 16) ? 0 : 8;
    const int sel16 = (lane < 16) ? 0 : 16;
    const int n     = lane & 15;

    const __bf16* arow0 = X + (size_t)(mbase + n) * DD;        // A rows m =  0..15
    const __bf16* arow1 = arow0 + (size_t)16 * DD;             // A rows m = 16..31
    const __bf16* brow  = W + (size_t)(nbase + n) * DD;        // B col n of tile t -> W row nbase+16t+n

    v8f acc[2][4];
#pragma unroll
    for (int mi = 0; mi < 2; ++mi)
#pragma unroll
        for (int t = 0; t < 4; ++t) acc[mi][t] = (v8f){};

    for (int k = 0; k < DD; k += 32) {
        BF16x16 a0, a1;
        a0.q[0] = *reinterpret_cast<const uint4*>(arow0 + k + sel8);
        a0.q[1] = *reinterpret_cast<const uint4*>(arow0 + k + 16 + sel8);
        a1.q[0] = *reinterpret_cast<const uint4*>(arow1 + k + sel8);
        a1.q[1] = *reinterpret_cast<const uint4*>(arow1 + k + 16 + sel8);
#pragma unroll
        for (int t = 0; t < 4; ++t) {
            const __bf16* br = brow + (size_t)(t * 16) * DD;
            BF16x16 bt;
            bt.q[0] = *reinterpret_cast<const uint4*>(br + k + sel8);
            bt.q[1] = *reinterpret_cast<const uint4*>(br + k + 16 + sel8);
            acc[0][t] = wmma_bf16(a0.v, bt.v, acc[0][t]);
            acc[1][t] = wmma_bf16(a1.v, bt.v, acc[1][t]);
        }
    }

#pragma unroll
    for (int t = 0; t < 4; ++t) {
        const int e  = nbase + t * 16 + n;
        const float bv = bias[e];
        const int h  = e >> 6;
        const int dv = e & 63;
#pragma unroll
        for (int mi = 0; mi < 2; ++mi) {
#pragma unroll
            for (int r = 0; r < 8; ++r) {
                int rm = mbase + mi * 16 + r + sel8;   // global row (b*S + s)
                int b_ = rm >> 11;
                int s_ = rm & (SS - 1);
                float vv = (acc[mi][t][r] + bv) * scale;
                if (mode == 0 || mode == 1) {
                    obf[(((size_t)b_ * HH + h) * SS + s_) * DVV + dv] = (__bf16)vv;
                } else if (mode == 2) {
                    obf[(((size_t)b_ * HH + h) * DVV + dv) * SS + s_] = (__bf16)vv;
                } else {
                    of[(size_t)rm * DD + e] = vv + resid[(size_t)rm * DD + e];
                }
            }
        }
    }
}

// ---------------------------------------------------------------- attention core
// One wave handles 16 query rows of one (b,h).
// Qh/Kh: bf16 [bh][s][dv]  (Q pre-scaled).  Vt: bf16 [bh][dv][s].
// att: f32 [bh][q][key] (normalized probs).  ctx: bf16 [b*S+q][h*64+dv].
__device__ __forceinline__ v8f score_tile(v16bf qa0, v16bf qa1,
                                          const __bf16* krow, int sel16) {
    BF16x16 kb0, kb1;
    kb0.q[0] = *reinterpret_cast<const uint4*>(krow + (sel16 >> 1));
    kb0.q[1] = *reinterpret_cast<const uint4*>(krow + 16 + (sel16 >> 1));
    kb1.q[0] = *reinterpret_cast<const uint4*>(krow + 32 + (sel16 >> 1));
    kb1.q[1] = *reinterpret_cast<const uint4*>(krow + 48 + (sel16 >> 1));
    v8f s = {};
    s = wmma_bf16(qa0, kb0.v, s);
    s = wmma_bf16(qa1, kb1.v, s);
    return s;
}

__global__ __launch_bounds__(32)
void attn_kernel(const __bf16* __restrict__ Qh, const __bf16* __restrict__ Kh,
                 const __bf16* __restrict__ Vt, const unsigned char* __restrict__ mask,
                 float* __restrict__ att, __bf16* __restrict__ ctx)
{
    const int lane  = threadIdx.x;
    const int qbase = blockIdx.x * 16;
    const int h     = blockIdx.y;
    const int b     = blockIdx.z;
    const int bh    = b * HH + h;
    const int sel8  = (lane < 16) ? 0 : 8;
    const int sel16 = (lane < 16) ? 0 : 16;
    const int n     = lane & 15;

    // Q A-fragments for dv 0..31 and 32..63
    const __bf16* qrow = Qh + ((size_t)bh * SS + qbase + n) * DVV;
    BF16x16 qa0, qa1;
    qa0.q[0] = *reinterpret_cast<const uint4*>(qrow + 0  + sel8);
    qa0.q[1] = *reinterpret_cast<const uint4*>(qrow + 16 + sel8);
    qa1.q[0] = *reinterpret_cast<const uint4*>(qrow + 32 + sel8);
    qa1.q[1] = *reinterpret_cast<const uint4*>(qrow + 48 + sel8);

    const unsigned char* mrow = mask + (size_t)b * SS * SS;
    const __bf16* kbase = Kh + (size_t)bh * SS * DVV;

    // ---- pass 1: row sums of exp(score) ----
    float rinv[8];
#pragma unroll
    for (int r = 0; r < 8; ++r) rinv[r] = 0.0f;

    for (int j = 0; j < SS / 16; ++j) {
        const __bf16* krow = kbase + (size_t)(j * 16 + n) * DVV;
        v8f s = score_tile(qa0.v, qa1.v, krow, sel16);
        int key = j * 16 + n;
#pragma unroll
        for (int r = 0; r < 8; ++r) {
            int qr = qbase + r + sel8;
            bool msk = mrow[(size_t)qr * SS + key] != 0;
            rinv[r] += msk ? 0.0f : __expf(s[r]);
        }
    }
#pragma unroll
    for (int r = 0; r < 8; ++r) {
        float v = rinv[r];
        for (int off = 8; off >= 1; off >>= 1) v += __shfl_xor(v, off, 16);
        rinv[r] = 1.0f / fmaxf(v, 1e-30f);
    }

    // ---- pass 2: write normalized probs, accumulate ctx = P * V ----
    __shared__ __align__(16) __bf16 Pl[16 * 32];
    v8f c0 = {}, c1 = {}, c2 = {}, c3 = {};
    const __bf16* vbase = Vt + (size_t)bh * DVV * SS;
    float* arow_base = att + (size_t)bh * SS * SS;

    for (int j2 = 0; j2 < SS / 32; ++j2) {
#pragma unroll
        for (int jj = 0; jj < 2; ++jj) {
            int j = j2 * 2 + jj;
            const __bf16* krow = kbase + (size_t)(j * 16 + n) * DVV;
            v8f s = score_tile(qa0.v, qa1.v, krow, sel16);
            int key = j * 16 + n;
#pragma unroll
            for (int r = 0; r < 8; ++r) {
                int qr = qbase + r + sel8;
                bool msk = mrow[(size_t)qr * SS + key] != 0;
                float p = msk ? 0.0f : __expf(s[r]) * rinv[r];
                arow_base[(size_t)qr * SS + key] = p;          // coalesced per row
                Pl[(r + sel8) * 32 + jj * 16 + n] = (__bf16)p; // C-layout -> LDS
            }
        }
        __syncthreads();
        // reload P as A-fragment (16 x 32 over the 32 keys of this step)
        BF16x16 pa;
        const __bf16* prow = &Pl[n * 32];
        pa.q[0] = *reinterpret_cast<const uint4*>(prow + sel8);
        pa.q[1] = *reinterpret_cast<const uint4*>(prow + 16 + sel8);
        // V B-fragments: column = dv tile*16 + n (a row of Vt), K = keys sel8..+7 and 16+sel8..+7
        size_t koff = (size_t)j2 * 32 + sel8;
#pragma unroll
        for (int t = 0; t < 4; ++t) {
            const __bf16* vr = vbase + (size_t)(t * 16 + n) * SS + koff;
            BF16x16 vb;
            vb.q[0] = *reinterpret_cast<const uint4*>(vr);
            vb.q[1] = *reinterpret_cast<const uint4*>(vr + 16);
            v8f& c = (t == 0) ? c0 : (t == 1) ? c1 : (t == 2) ? c2 : c3;
            c = wmma_bf16(pa.v, vb.v, c);
        }
        __syncthreads();
    }

    // store ctx (row-major [b*S+q][D]) as bf16 for the output GEMM
    const int col0 = h * DVV + n;
#pragma unroll
    for (int r = 0; r < 8; ++r) {
        size_t g = (size_t)(b * SS + qbase + r + sel8) * DD + col0;
        ctx[g + 0]  = (__bf16)c0[r];
        ctx[g + 16] = (__bf16)c1[r];
        ctx[g + 32] = (__bf16)c2[r];
        ctx[g + 48] = (__bf16)c3[r];
    }
}

// ---------------------------------------------------------------- layernorm
__global__ __launch_bounds__(256)
void layernorm_k(const float* __restrict__ x, const float* __restrict__ g,
                 const float* __restrict__ bta, float* __restrict__ y)
{
    int row = blockIdx.x;
    const float* xr = x + (size_t)row * DD;
    float s = 0.f, s2 = 0.f;
    for (int i = threadIdx.x; i < DD; i += 256) { float v = xr[i]; s += v; s2 += v * v; }
    __shared__ float red0[256], red1[256];
    red0[threadIdx.x] = s; red1[threadIdx.x] = s2;
    __syncthreads();
    for (int o = 128; o >= 1; o >>= 1) {
        if (threadIdx.x < o) { red0[threadIdx.x] += red0[threadIdx.x + o];
                               red1[threadIdx.x] += red1[threadIdx.x + o]; }
        __syncthreads();
    }
    float mean = red0[0] * (1.0f / DD);
    float var  = red1[0] * (1.0f / DD) - mean * mean;
    float inv  = rsqrtf(var + LNEPS);
    for (int i = threadIdx.x; i < DD; i += 256)
        y[(size_t)row * DD + i] = (xr[i] - mean) * inv * g[i] + bta[i];
}

// ---------------------------------------------------------------- launch
extern "C" void kernel_launch(void* const* d_in, const int* in_sizes, int n_in,
                              void* d_out, int out_size, void* d_ws, size_t ws_size,
                              hipStream_t stream) {
    const float* q    = (const float*)d_in[0];
    const float* k    = (const float*)d_in[1];
    const float* v    = (const float*)d_in[2];
    const unsigned char* mask = (const unsigned char*)d_in[3];
    const float* Wq = (const float*)d_in[4];  const float* bq = (const float*)d_in[5];
    const float* Wk = (const float*)d_in[6];  const float* bk = (const float*)d_in[7];
    const float* Wv = (const float*)d_in[8];  const float* bv = (const float*)d_in[9];
    const float* Wo = (const float*)d_in[10]; const float* bo = (const float*)d_in[11];
    const float* ln_g = (const float*)d_in[12];
    const float* ln_b = (const float*)d_in[13];

    const size_t MB = 1024ull * 1024ull;
    char* ws = (char*)d_ws;
    __bf16* qb  = (__bf16*)(ws + 0 * MB);    // 8MB each (4M bf16)
    __bf16* kb  = (__bf16*)(ws + 8 * MB);
    __bf16* vb  = (__bf16*)(ws + 16 * MB);
    __bf16* Wqb = (__bf16*)(ws + 24 * MB);   // 2MB each
    __bf16* Wkb = (__bf16*)(ws + 26 * MB);
    __bf16* Wvb = (__bf16*)(ws + 28 * MB);
    __bf16* Wob = (__bf16*)(ws + 30 * MB);
    __bf16* Qh  = (__bf16*)(ws + 32 * MB);   // 8MB each
    __bf16* Kh  = (__bf16*)(ws + 40 * MB);
    __bf16* Vt  = (__bf16*)(ws + 48 * MB);
    __bf16* ctx = (__bf16*)(ws + 56 * MB);
    float*  xr  = (float*) (ws + 64 * MB);   // 16MB

    float* y   = (float*)d_out;
    float* att = y + (size_t)BB * SS * DD;

    const int nx8 = (BB * SS * DD) / 8;   // 512K vec8 groups
    const int nw8 = (DD * DD) / 8;        // 128K vec8 groups
    cvt_bf16_v<<<(nx8 + 255) / 256, 256, 0, stream>>>((const float4*)q, (uint4*)qb, nx8);
    cvt_bf16_v<<<(nx8 + 255) / 256, 256, 0, stream>>>((const float4*)k, (uint4*)kb, nx8);
    cvt_bf16_v<<<(nx8 + 255) / 256, 256, 0, stream>>>((const float4*)v, (uint4*)vb, nx8);
    cvt_bf16_v<<<(nw8 + 255) / 256, 256, 0, stream>>>((const float4*)Wq, (uint4*)Wqb, nw8);
    cvt_bf16_v<<<(nw8 + 255) / 256, 256, 0, stream>>>((const float4*)Wk, (uint4*)Wkb, nw8);
    cvt_bf16_v<<<(nw8 + 255) / 256, 256, 0, stream>>>((const float4*)Wv, (uint4*)Wvb, nw8);
    cvt_bf16_v<<<(nw8 + 255) / 256, 256, 0, stream>>>((const float4*)Wo, (uint4*)Wob, nw8);

    const float scaleQ = 1.0f / sqrtf((float)SS);   // faithful to reference "bug"
    dim3 gg(DD / 64, (BB * SS) / 32);
    proj_gemm<<<gg, 32, 0, stream>>>(qb, Wqb, bq, scaleQ, 0, Qh, nullptr, nullptr);
    proj_gemm<<<gg, 32, 0, stream>>>(kb, Wkb, bk, 1.0f,   1, Kh, nullptr, nullptr);
    proj_gemm<<<gg, 32, 0, stream>>>(vb, Wvb, bv, 1.0f,   2, Vt, nullptr, nullptr);

    attn_kernel<<<dim3(SS / 16, HH, BB), 32, 0, stream>>>(Qh, Kh, Vt, mask, att, ctx);

    proj_gemm<<<gg, 32, 0, stream>>>(ctx, Wob, bo, 1.0f, 3, nullptr, xr, q);

    layernorm_k<<<BB * SS, 256, 0, stream>>>(xr, ln_g, ln_b, y);
}
